// MeanShift_28381143892902
// MI455X (gfx1250) — matrix operand order repaired and checked by reading, not compiled, not run on hardware
//
#include <hip/hip_runtime.h>
#include <hip/hip_bf16.h>

// ---------------------------------------------------------------------------
// Problem constants (match reference setup_inputs)
// ---------------------------------------------------------------------------
#define BATCH 256
#define CDIM  512
#define KROWS 128000
#define NTILE 16          // bank rows per WMMA tile
#define KC    16          // K-chunks of 32 along CDIM (512/32)
#define NT_PER_WG 25      // bank tiles per workgroup (N direction)
#define NGROUPS  320      // 8000 tiles / 25
#define WAVES 4           // waves per workgroup in gemm kernel
#define TOPK 5

typedef __attribute__((ext_vector_type(16))) __bf16 v16bf;
typedef __attribute__((ext_vector_type(8)))  float  v8f;

union VB {
  v16bf v;
  uint4 q[2];
};

struct SMemTile { unsigned short hi[NTILE * CDIM]; unsigned short lo[NTILE * CDIM]; }; // 32 KB
struct SMemMerge { float c[WAVES * 32 * 40]; unsigned int i[WAVES * 32 * 40]; };        // 40 KB
union SMemU { SMemTile tile; SMemMerge mg; };

// ---------------------------------------------------------------------------
// Helpers
// ---------------------------------------------------------------------------
__device__ __forceinline__ unsigned short bf16_rne(float f) {
  unsigned u = __float_as_uint(f);
  return (unsigned short)((u + 0x7FFFu + ((u >> 16) & 1u)) >> 16);
}
__device__ __forceinline__ float bf16_to_f(unsigned short h) {
  return __uint_as_float(((unsigned)h) << 16);
}
__device__ __forceinline__ void split2(float x, unsigned short& h, unsigned short& l) {
  h = bf16_rne(x);
  l = bf16_rne(x - bf16_to_f(h));
}
// keep the 5 LARGEST keys, sorted descending (branchless insertion)
__device__ __forceinline__ void ins5(float c, unsigned i, float* kb, unsigned* ib) {
#pragma unroll
  for (int j = 0; j < TOPK; ++j) {
    bool gt = c > kb[j];
    float tc = kb[j]; unsigned ti = ib[j];
    kb[j] = gt ? c : kb[j]; ib[j] = gt ? i : ib[j];
    c = gt ? tc : c;        i = gt ? ti : i;
  }
}

// ---------------------------------------------------------------------------
// Kernel 1: normalize current_target -> bf16 hi/lo A matrix; query -> qhat f32
// grid 512 blocks x 256 threads (block b<256: t row b ; else q row b-256)
// ---------------------------------------------------------------------------
__global__ void ms_prep_kernel(const float* __restrict__ qin, const float* __restrict__ tin,
                               unsigned short* __restrict__ aHi, unsigned short* __restrict__ aLo,
                               float* __restrict__ qhat) {
  __shared__ float red[256];
  const int b = blockIdx.x, tid = threadIdx.x;
  const bool isT = b < BATCH;
  const float* src = isT ? (tin + (size_t)b * CDIM) : (qin + (size_t)(b - BATCH) * CDIM);
  const float x0 = src[tid], x1 = src[tid + 256];
  red[tid] = x0 * x0 + x1 * x1;
  __syncthreads();
  for (int s = 128; s > 0; s >>= 1) {
    if (tid < s) red[tid] += red[tid + s];
    __syncthreads();
  }
  const float rn = 1.0f / fmaxf(sqrtf(red[0]), 1e-12f);
  const float y0 = x0 * rn, y1 = x1 * rn;
  if (isT) {
    const size_t o = (size_t)b * CDIM;
    unsigned short h0, l0, h1, l1;
    split2(y0, h0, l0);
    split2(y1, h1, l1);
    aHi[o + tid] = h0; aHi[o + tid + 256] = h1;
    aLo[o + tid] = l0; aLo[o + tid + 256] = l1;
  } else {
    const size_t o = (size_t)(b - BATCH) * CDIM;
    qhat[o + tid] = y0; qhat[o + tid + 256] = y1;
  }
}

// ---------------------------------------------------------------------------
// Kernel 2: rnorm[k] = 1/max(||bank_k||, eps). 16000 blocks x 256 (wave/row)
// ---------------------------------------------------------------------------
__global__ void ms_bank_norm_kernel(const float* __restrict__ bank, float* __restrict__ rnorm) {
  const int wave = threadIdx.x >> 5, lane = threadIdx.x & 31;
  const int row = blockIdx.x * 8 + wave;
  const float* p = bank + (size_t)row * CDIM;
  float ss = 0.0f;
#pragma unroll
  for (int i = 0; i < 4; ++i) {
    float4 v = *(const float4*)(p + lane * 4 + i * 128);
    ss += v.x * v.x + v.y * v.y + v.z * v.z + v.w * v.w;
  }
#pragma unroll
  for (int m = 16; m >= 1; m >>= 1) ss += __shfl_xor(ss, m);
  if (lane == 0) rnorm[row] = 1.0f / fmaxf(sqrtf(ss), 1e-12f);
}

// ---------------------------------------------------------------------------
// Kernel 3: fused GEMM (bf16 split, 3 WMMAs per K-chunk) + running top-5.
// grid (NGROUPS, 4), block 128 (4 waves). Wave owns M-tile of 16 batch rows,
// keeps its A fragments (16 K-chunks, hi+lo) resident in VGPRs.
// Bank tile -> LDS once per WG (coop load + fp32->bf16 hi/lo split).
// ---------------------------------------------------------------------------
__global__ void __launch_bounds__(128) ms_gemm_topk_kernel(
    const unsigned short* __restrict__ aHiG, const unsigned short* __restrict__ aLoG,
    const float* __restrict__ bank, const float* __restrict__ rnorm,
    uint2* __restrict__ partial) {
  __shared__ SMemU smem;
  const int tid = threadIdx.x;
  const int wave = tid >> 5, lane = tid & 31;
  const int mbase = (blockIdx.y * WAVES + wave) * 16;

  // --- A fragments resident in registers: ISA 16-bit A layout (16x32) ---
  // lane<16: row M=lane, K = {0..7, 16..23} of chunk; lane>=16: row M=lane-16, K={8..15,24..31}
  VB afHi[KC], afLo[KC];
  {
    const int arow = mbase + (lane & 15);
    const int half = lane >> 4;
#pragma unroll
    for (int kc = 0; kc < KC; ++kc) {
      const int c0 = kc * 32 + half * 8;
      const unsigned short* ph = aHiG + arow * CDIM + c0;
      const unsigned short* pl = aLoG + arow * CDIM + c0;
      afHi[kc].q[0] = *(const uint4*)ph;
      afHi[kc].q[1] = *(const uint4*)(ph + 16);
      afLo[kc].q[0] = *(const uint4*)pl;
      afLo[kc].q[1] = *(const uint4*)(pl + 16);
    }
  }

  // per-lane running top-5 per C/D register row (8 rows per lane)
  float kb[8][TOPK]; unsigned ib[8][TOPK];
#pragma unroll
  for (int r = 0; r < 8; ++r)
#pragma unroll
    for (int j = 0; j < TOPK; ++j) { kb[r][j] = -3.4e38f; ib[r][j] = 0u; }

  const int nt0 = blockIdx.x * NT_PER_WG;
  for (int ntl = 0; ntl < NT_PER_WG; ++ntl) {
    const int nbase = (nt0 + ntl) * NTILE;

    // --- cooperative load + bf16-split of bank tile into LDS ---
#pragma unroll
    for (int i = 0; i < 16; ++i) {
      const int e = (tid + i * 128) * 4;          // element index in [0, 8192)
      const int r = e >> 9, c = e & (CDIM - 1);
      float4 v = *(const float4*)(bank + (size_t)(nbase + r) * CDIM + c);
      unsigned short h0, l0, h1, l1, h2, l2, h3, l3;
      split2(v.x, h0, l0); split2(v.y, h1, l1);
      split2(v.z, h2, l2); split2(v.w, h3, l3);
      *(uint2*)&smem.tile.hi[e] =
          make_uint2((unsigned)h0 | ((unsigned)h1 << 16), (unsigned)h2 | ((unsigned)h3 << 16));
      *(uint2*)&smem.tile.lo[e] =
          make_uint2((unsigned)l0 | ((unsigned)l1 << 16), (unsigned)l2 | ((unsigned)l3 << 16));
    }
    __syncthreads();

    const float rn = rnorm[nbase + (lane & 15)];
    v8f acc = {};
    // B layout (32x16, 16-bit): lane<16 -> N=lane, K={0..7,16..23};
    // lane>=16 -> N=lane-16, K={8..15,24..31}.
#pragma unroll
    for (int kc = 0; kc < KC; ++kc) {
      const int boff = (lane & 15) * CDIM + kc * 32 + (lane >> 4) * 8;
      VB bh, bl;
      bh.q[0] = *(const uint4*)&smem.tile.hi[boff];
      bh.q[1] = *(const uint4*)&smem.tile.hi[boff + 16];
      bl.q[0] = *(const uint4*)&smem.tile.lo[boff];
      bl.q[1] = *(const uint4*)&smem.tile.lo[boff + 16];
      acc = __builtin_amdgcn_wmma_f32_16x16x32_bf16(false, afHi[kc].v, false, bh.v, (short)0, acc, false, false);
      acc = __builtin_amdgcn_wmma_f32_16x16x32_bf16(false, afLo[kc].v, false, bh.v, (short)0, acc, false, false);
      acc = __builtin_amdgcn_wmma_f32_16x16x32_bf16(false, afHi[kc].v, false, bl.v, (short)0, acc, false, false);
    }

    // candidates: cos = s * rnorm; larger cos == smaller dist_t
    const unsigned nidx = (unsigned)(nbase + (lane & 15));
#pragma unroll
    for (int r = 0; r < 8; ++r) ins5(acc[r] * rn, nidx, kb[r], ib[r]);
    __syncthreads();
  }

  // --- intra-wave merge via LDS (reuse tile storage) ---
  const int dbase = (wave * 32 + lane) * 40;
#pragma unroll
  for (int r = 0; r < 8; ++r)
#pragma unroll
    for (int j = 0; j < TOPK; ++j) {
      smem.mg.c[dbase + r * TOPK + j] = kb[r][j];
      smem.mg.i[dbase + r * TOPK + j] = ib[r][j];
    }
  __syncthreads();
  if (lane < 16) {
    const int rslot = lane & 7;
    const int lstart = (lane < 8) ? 0 : 16;
    float mk[TOPK]; unsigned mi[TOPK];
#pragma unroll
    for (int j = 0; j < TOPK; ++j) { mk[j] = -3.4e38f; mi[j] = 0u; }
    for (int sl = 0; sl < 16; ++sl) {
      const int sb = (wave * 32 + lstart + sl) * 40 + rslot * TOPK;
#pragma unroll
      for (int j = 0; j < TOPK; ++j) ins5(smem.mg.c[sb + j], smem.mg.i[sb + j], mk, mi);
    }
    const int row = mbase + lane;
    uint2* out = partial + ((size_t)row * NGROUPS + blockIdx.x) * TOPK;
#pragma unroll
    for (int j = 0; j < TOPK; ++j) out[j] = make_uint2(__float_as_uint(mk[j]), mi[j]);
  }
}

// ---------------------------------------------------------------------------
// Kernel 4: per-row merge of NGROUPS partials; recompute dist_q at top-5
// indices; loss/purity per row. grid 256 x 32 (one wave per row)
// ---------------------------------------------------------------------------
__global__ void ms_finalize_rows_kernel(
    const float* __restrict__ qhat, const float* __restrict__ bank,
    const float* __restrict__ rnorm, const uint2* __restrict__ partial,
    const int* __restrict__ labels, const int* __restrict__ labq,
    float* __restrict__ rowLoss, float* __restrict__ rowPur) {
  const int row = blockIdx.x;
  const int lane = threadIdx.x;
  float mk[TOPK]; unsigned mi[TOPK];
#pragma unroll
  for (int j = 0; j < TOPK; ++j) { mk[j] = -3.4e38f; mi[j] = 0u; }
  for (int g = lane; g < NGROUPS; g += 32) {
    const uint2* p = partial + ((size_t)row * NGROUPS + g) * TOPK;
#pragma unroll
    for (int j = 0; j < TOPK; ++j) ins5(__uint_as_float(p[j].x), p[j].y, mk, mi);
  }
  __shared__ float sc[32 * TOPK];
  __shared__ unsigned si[32 * TOPK];
  __shared__ unsigned topIdx[TOPK];
#pragma unroll
  for (int j = 0; j < TOPK; ++j) { sc[lane * TOPK + j] = mk[j]; si[lane * TOPK + j] = mi[j]; }
  __syncthreads();
  if (lane == 0) {
    float fk[TOPK]; unsigned fi[TOPK];
#pragma unroll
    for (int j = 0; j < TOPK; ++j) { fk[j] = -3.4e38f; fi[j] = 0u; }
    for (int s = 0; s < 32 * TOPK; ++s) ins5(sc[s], si[s], fk, fi);
#pragma unroll
    for (int j = 0; j < TOPK; ++j) topIdx[j] = fi[j];
  }
  __syncthreads();

  float lsum = 0.0f, psum = 0.0f;
  const int lab = labels[row];
  const float* qp = qhat + (size_t)row * CDIM;
#pragma unroll
  for (int j = 0; j < TOPK; ++j) {
    const unsigned id = topIdx[j];
    const float* bp = bank + (size_t)id * CDIM;
    float dot = 0.0f;
#pragma unroll
    for (int i = 0; i < 4; ++i) {
      float4 a = *(const float4*)(qp + lane * 4 + i * 128);
      float4 b = *(const float4*)(bp + lane * 4 + i * 128);
      dot += a.x * b.x + a.y * b.y + a.z * b.z + a.w * b.w;
    }
#pragma unroll
    for (int m = 16; m >= 1; m >>= 1) dot += __shfl_xor(dot, m);
    lsum += 2.0f - 2.0f * (dot * rnorm[id]);
    psum += (labq[id] == lab) ? 1.0f : 0.0f;
  }
  if (lane == 0) { rowLoss[row] = lsum * 0.2f; rowPur[row] = psum * 0.2f; }
}

// ---------------------------------------------------------------------------
// Kernel 5: exact deterministic reduction over 256 rows -> d_out[0..1]
// ---------------------------------------------------------------------------
__global__ void ms_final_reduce_kernel(const float* __restrict__ rowLoss,
                                       const float* __restrict__ rowPur,
                                       float* __restrict__ out) {
  __shared__ float s1[256], s2[256];
  const int tid = threadIdx.x;
  s1[tid] = rowLoss[tid];
  s2[tid] = rowPur[tid];
  __syncthreads();
  for (int s = 128; s > 0; s >>= 1) {
    if (tid < s) { s1[tid] += s1[tid + s]; s2[tid] += s2[tid + s]; }
    __syncthreads();
  }
  if (tid == 0) {
    out[0] = s1[0] * (1.0f / 256.0f);
    out[1] = s2[0] * (1.0f / 256.0f);
  }
}

// ---------------------------------------------------------------------------
// Launch
// ---------------------------------------------------------------------------
extern "C" void kernel_launch(void* const* d_in, const int* in_sizes, int n_in,
                              void* d_out, int out_size, void* d_ws, size_t ws_size,
                              hipStream_t stream) {
  const float* q    = (const float*)d_in[0];
  const float* t    = (const float*)d_in[1];
  const float* bank = (const float*)d_in[2];
  const int*   labels = (const int*)d_in[3];
  const int*   labq   = (const int*)d_in[4];

  char* ws = (char*)d_ws;
  unsigned short* aHi  = (unsigned short*)(ws + 0);          // 256*512*2  = 262144 B
  unsigned short* aLo  = (unsigned short*)(ws + 262144);     // 262144 B
  float* qhat          = (float*)(ws + 524288);              // 256*512*4  = 524288 B
  float* rnorm         = (float*)(ws + 1048576);             // 128000*4   = 512000 B
  uint2* partial       = (uint2*)(ws + 1572864);             // 256*320*5*8 = 3276800 B
  float* rowLoss       = (float*)(ws + 4849664);             // 1024 B
  float* rowPur        = (float*)(ws + 4850688);             // 1024 B  (total ~4.63 MB)

  ms_prep_kernel<<<512, 256, 0, stream>>>(q, t, aHi, aLo, qhat);
  ms_bank_norm_kernel<<<KROWS / 8, 256, 0, stream>>>(bank, rnorm);
  ms_gemm_topk_kernel<<<dim3(NGROUPS, 4), 128, 0, stream>>>(aHi, aLo, bank, rnorm, partial);
  ms_finalize_rows_kernel<<<BATCH, 32, 0, stream>>>(qhat, bank, rnorm, partial, labels, labq,
                                                    rowLoss, rowPur);
  ms_final_reduce_kernel<<<1, 256, 0, stream>>>(rowLoss, rowPur, (float*)d_out);
}
